// V7AstrometryMatcher_83485574299693
// MI455X (gfx1250) — matrix operand order repaired and checked by reading, not compiled, not run on hardware
//
#include <hip/hip_runtime.h>
#include <hip/hip_bf16.h>
#include <math.h>
#include <stdint.h>

// ---- Problem constants (from reference) ----
#define BB   64
#define CC   64
#define HH   64
#define WW   64
#define HWW  (HH*WW)           // 4096
#define KK   49                // 7x7 offsets
#define RR   3

typedef __attribute__((ext_vector_type(16))) _Float16 v16h;
typedef __attribute__((ext_vector_type(8)))  _Float16 v8h;
typedef __attribute__((ext_vector_type(8)))  float    v8f;
typedef __attribute__((ext_vector_type(4)))  unsigned int u32x4;
typedef __attribute__((ext_vector_type(8)))  int      i32x8;
typedef __attribute__((ext_vector_type(4)))  int      i32x4;

#define WMMA16(a,b,c) __builtin_amdgcn_wmma_f32_16x16x32_f16(false,(a),false,(b),(short)0,(c),false,false)

#if __has_builtin(__builtin_amdgcn_tensor_load_to_lds)
#define HAVE_TDM 1
#else
#define HAVE_TDM 0
#endif

__device__ __forceinline__ float wave_red(float v) {
    #pragma unroll
    for (int off = 16; off; off >>= 1) v += __shfl_xor(v, off, 32);
    return v;
}

__device__ __forceinline__ float gelu_exact(float x) {
    return 0.5f * x * (1.0f + erff(x * 0.70710678118654752f));
}

#if HAVE_TDM
// ---------------------------------------------------------------------------
// Build a 2-D Tensor Data Mover descriptor (D#) and issue TENSOR_LOAD_TO_LDS.
// dim0 = contiguous elements (pixels), dim1 = strided rows (channels).
// Group0: count=1 | lds_addr | global_addr[56:0] | type=2 ("image").
// Group1: wg_mask=0, data_size=4B, tensor_dim0/1, tile_dim0/1, dim0-stride.
// Groups 2/3: zero (<=2D tensor). D# semantics per CDNA5 ISA.
// ---------------------------------------------------------------------------
__device__ __forceinline__ void tdm_load_2d(unsigned lds_off, const void* gptr,
                                            unsigned tile_d0, unsigned tile_d1,
                                            unsigned tensor_d0, unsigned tensor_d1,
                                            unsigned long long stride0_elts) {
    unsigned long long ga = (unsigned long long)(uintptr_t)gptr;
    u32x4 g0;
    g0[0] = 1u;                                           // count=1 (valid), user mode
    g0[1] = lds_off;                                      // LDS byte address
    g0[2] = (unsigned)(ga & 0xffffffffull);               // global_addr[31:0]
    g0[3] = (unsigned)((ga >> 32) & 0x1ffffffull)         // global_addr[56:32]
          | (2u << 30);                                   // type=2
    i32x8 g1;
    g1[0] = (int)(2u << 16);                              // data_size code 2 = 4 bytes
    g1[1] = (int)((tensor_d0 & 0xffffu) << 16);           // tensor_dim0[15:0] @ bits 63:48
    g1[2] = (int)((tensor_d0 >> 16) | ((tensor_d1 & 0xffffu) << 16));
    g1[3] = (int)((tensor_d1 >> 16) | (tile_d0 << 16));   // tile_dim0 @ bits 127:112
    g1[4] = (int)(tile_d1 & 0xffffu);                     // tile_dim1; tile_dim2=0
    g1[5] = (int)(unsigned)(stride0_elts & 0xffffffffull);
    g1[6] = (int)(unsigned)((stride0_elts >> 32) & 0xffffull);
    g1[7] = 0;
    i32x4 z4 = {0, 0, 0, 0};
#if defined(__clang_major__) && __clang_major__ >= 23
    i32x8 z8 = {0, 0, 0, 0, 0, 0, 0, 0};
    __builtin_amdgcn_tensor_load_to_lds(g0, g1, z4, z4, z8, 0);
#else
    __builtin_amdgcn_tensor_load_to_lds(g0, g1, z4, z4, 0);
#endif
}
#endif

// ---------------------------------------------------------------------------
// Kernel 0: init. Compute Gaussian normalizer G = (sum exp(-2 y^2))^2 and
// zero the per-batch accumulators (sumEG, logits).
// ---------------------------------------------------------------------------
__global__ __launch_bounds__(64) void k_init(float* gsum, float* sumEG, float* logits) {
    __shared__ float red[64];
    int t = threadIdx.x;
    float y = -1.0f + 2.0f * (float)t / 63.0f;
    red[t] = expf(-2.0f * y * y);
    __syncthreads();
    #pragma unroll
    for (int s = 32; s > 0; s >>= 1) { if (t < s) red[t] += red[t + s]; __syncthreads(); }
    if (t == 0) gsum[0] = red[0] * red[0];
    sumEG[t] = 0.0f;
    for (int i = t; i < BB * KK; i += 64) logits[i] = 0.0f;
}

// ---------------------------------------------------------------------------
// Kernel 1: 1x1-conv projection (WMMA GEMM) + per-pixel channel L2 normalize.
// The workgroup's 128-pixel x 64-channel fp32 feature tiles (rubin + vis) are
// staged into LDS by the Tensor Data Mover (double-buffered: both descriptors
// issued up-front, s_wait_tensorcnt 1 gates rubin, 0 gates vis, so the vis
// DMA overlaps the rubin WMMA work). Each wave then computes
// out[64ch x 16pix] = W[64x64] * feat[64 x 16pix] via 4 M-tiles x 2 K-chunks
// of v_wmma_f32_16x16x32_f16, L2-normalizes per pixel, stores f16 features
// [pixel][channel], and accumulates sumEG[b] = sum(E * gauss_norm).
// ---------------------------------------------------------------------------
__global__ __launch_bounds__(256) void k_proj(const float* __restrict__ rfeat,
                                              const float* __restrict__ vfeat,
                                              const float* __restrict__ Wr,
                                              const float* __restrict__ Wv,
                                              const float* __restrict__ gsumP,
                                              _Float16* __restrict__ rn,
                                              _Float16* __restrict__ vn,
                                              float* __restrict__ energy,
                                              float* __restrict__ sumEG) {
    __shared__ float sfeatR[CC * 128];   // 32 KB: [channel][pixel-in-WG]
    __shared__ float sfeatV[CC * 128];   // 32 KB
    const int wave = threadIdx.x >> 5;
    const int lane = threadIdx.x & 31;
    const int hi   = lane >> 4;
    const int nloc = lane & 15;
    const int b      = blockIdx.x >> 5;        // 32 WGs per batch
    const int wgInB  = blockIdx.x & 31;
    const int pWG    = wgInB * 128;            // first pixel of this WG
    const int pLoc   = wave * 16 + nloc;       // pixel-in-WG for this lane
    const int pix    = pWG + pLoc;
    const float G    = gsumP[0];

    const float* gR = rfeat + (size_t)b * CC * HWW + pWG;
    const float* gV = vfeat + (size_t)b * CC * HWW + pWG;

#if HAVE_TDM
    if (wave == 0) {
        tdm_load_2d((unsigned)(uintptr_t)(void*)sfeatR, gR, 128u, (unsigned)CC,
                    (unsigned)HWW, (unsigned)CC, (unsigned long long)HWW);
        tdm_load_2d((unsigned)(uintptr_t)(void*)sfeatV, gV, 128u, (unsigned)CC,
                    (unsigned)HWW, (unsigned)CC, (unsigned long long)HWW);
        __builtin_amdgcn_s_wait_tensorcnt(1);  // rubin tile landed (in-order per wave)
    }
    // speculative prefetch of the vis weights while rubin is processed
    __builtin_prefetch(&Wv[threadIdx.x], 0, 3);
#else
    for (int i = threadIdx.x; i < CC * 128; i += 256) {
        sfeatR[i] = gR[(size_t)(i >> 7) * HWW + (i & 127)];
        sfeatV[i] = gV[(size_t)(i >> 7) * HWW + (i & 127)];
    }
#endif
    __syncthreads();

    for (int m = 0; m < 2; ++m) {
        if (m == 1) {
#if HAVE_TDM
            if (wave == 0) __builtin_amdgcn_s_wait_tensorcnt(0);  // vis tile landed
#endif
            __syncthreads();
        }
        const float* sf = m ? sfeatV : sfeatR;
        const float* Wm = m ? Wv     : Wr;

        // B matrix from LDS. table order: K=c0+8*hi+j (j<8), c0+16+8*hi+(j-8) (j>=8)
        v16h bv[2];
        #pragma unroll
        for (int kc = 0; kc < 2; ++kc) {
            const int c0 = kc * 32;
            #pragma unroll
            for (int j = 0; j < 16; ++j) {
                int k = c0 + (j < 8 ? hi * 8 + j : 16 + hi * 8 + (j - 8));
                bv[kc][j] = (_Float16)sf[k * 128 + pLoc];
            }
        }

        v8f acc[4];
        #pragma unroll
        for (int ot = 0; ot < 4; ++ot) {
            v8f c = {};
            #pragma unroll
            for (int kc = 0; kc < 2; ++kc) {
                const int c0 = kc * 32;
                const int row = ot * 16 + nloc;   // output channel row
                v16h a;
                #pragma unroll
                for (int e = 0; e < 8; ++e) a[e]     = (_Float16)Wm[row * CC + c0 + hi * 8 + e];
                #pragma unroll
                for (int e = 0; e < 8; ++e) a[8 + e] = (_Float16)Wm[row * CC + c0 + 16 + hi * 8 + e];
                c = WMMA16(a, bv[kc], c);
            }
            acc[ot] = c;
        }

        // Channel L2 norm for this lane's pixel (lane holds 32 of 64 channels;
        // partner lane (xor 16) holds the other 32).
        float s = 0.0f;
        #pragma unroll
        for (int ot = 0; ot < 4; ++ot)
            #pragma unroll
            for (int r = 0; r < 8; ++r) s += acc[ot][r] * acc[ot][r];
        s += __shfl_xor(s, 16, 32);
        const float inv = 1.0f / fmaxf(sqrtf(s), 1e-6f);

        _Float16* dst = m ? vn : rn;
        const size_t pbase = ((size_t)(b * HWW + pix)) * CC;
        #pragma unroll
        for (int ot = 0; ot < 4; ++ot) {
            v8h hv;
            #pragma unroll
            for (int r = 0; r < 8; ++r) hv[r] = (_Float16)(acc[ot][r] * inv);
            *(v8h*)(dst + pbase + ot * 16 + hi * 8) = hv;
        }

        if (m == 0) {
            const float E = s * inv * inv;   // ||rn||^2 after normalization
            if (hi == 0) energy[b * HWW + pix] = E;
            const int hh = pix >> 6, wwp = pix & 63;
            const float yy = -1.0f + 2.0f * (float)hh  / 63.0f;
            const float xx = -1.0f + 2.0f * (float)wwp / 63.0f;
            const float g  = expf(-2.0f * (yy * yy + xx * xx)) / G;
            float eg = (hi == 0) ? E * g : 0.0f;
            eg = wave_red(eg);
            if (lane == 0) atomicAdd(&sumEG[b], eg);
        }
    }
}

// ---------------------------------------------------------------------------
// Kernel 2: mean-pool the ORIGINAL features over H,W -> [B,C].
// ---------------------------------------------------------------------------
__global__ __launch_bounds__(128) void k_pool(const float* __restrict__ rfeat,
                                              const float* __restrict__ vfeat,
                                              float* __restrict__ rpool,
                                              float* __restrict__ vpool) {
    const int which = blockIdx.x >> 12;       // 4096 (b,c) pairs per input
    const int idx   = blockIdx.x & 4095;
    const float* src = which ? vfeat : rfeat;
    float s = 0.0f;
    for (int i = threadIdx.x; i < HWW; i += 128) s += src[(size_t)idx * HWW + i];
    __shared__ float red[128];
    red[threadIdx.x] = s;
    __syncthreads();
    #pragma unroll
    for (int st = 64; st > 0; st >>= 1) { if (threadIdx.x < st) red[threadIdx.x] += red[threadIdx.x + st]; __syncthreads(); }
    if (threadIdx.x == 0) (which ? vpool : rpool)[idx] = red[0] * (1.0f / (float)HWW);
}

// ---------------------------------------------------------------------------
// Kernel 3: 49-offset cost volume via WMMA.
// Wave = one row-tile (b, h, w0..w0+15). For each dy (7) and B-window (2),
// compute a 16x16 pixel-pixel dot tile (2 K-chunk WMMAs); replicate padding
// = column/row index clamping. Harvest the dx = n - m - 3 + 16*bt diagonal
// band weighted by (spatial_w + 1e-10) through LDS (no dynamic VGPR index),
// accumulate into per-WG logits, one global atomic flush.
// ---------------------------------------------------------------------------
__global__ __launch_bounds__(256) void k_corr(const _Float16* __restrict__ rn,
                                              const _Float16* __restrict__ vn,
                                              const float* __restrict__ energy,
                                              const float* __restrict__ sumEG,
                                              const float* __restrict__ gsumP,
                                              float* __restrict__ logits) {
    __shared__ float sTile[8][16 * 17];
    __shared__ float sWgt[8][16];
    __shared__ float sLog[KK];

    const int wave = threadIdx.x >> 5;
    const int lane = threadIdx.x & 31;
    const int hi   = lane >> 4;
    const int nloc = lane & 15;

    if (threadIdx.x < KK) sLog[threadIdx.x] = 0.0f;
    __syncthreads();

    const int t   = blockIdx.x * 8 + wave;   // row-tile id
    const int b   = t >> 8;                  // 256 tiles per batch (WG stays in one b)
    const int rem = t & 255;
    const int h   = rem >> 2;
    const int w0  = (rem & 3) << 4;

    // A = rn[16 pixels x 64 ch], two 16x32 K-chunks resident.
    const size_t baseA = ((size_t)(b * HWW + h * WW + w0 + nloc)) * CC;
    v8h a00 = *(const v8h*)(rn + baseA +  0 + hi * 16);
    v8h a01 = *(const v8h*)(rn + baseA +  8 + hi * 16);
    v8h a10 = *(const v8h*)(rn + baseA + 32 + hi * 16);
    v8h a11 = *(const v8h*)(rn + baseA + 40 + hi * 16);
    v16h a0 = __builtin_shufflevector(a00, a01, 0,1,2,3,4,5,6,7,8,9,10,11,12,13,14,15);
    v16h a1 = __builtin_shufflevector(a10, a11, 0,1,2,3,4,5,6,7,8,9,10,11,12,13,14,15);

    // Per-pixel spatial weights: wgt = E*g/(sumEG+1e-8) + 1e-10.
    const float invS = 1.0f / (sumEG[b] + 1e-8f);
    const float G    = gsumP[0];
    if (hi == 0) {
        const int wp = w0 + nloc;
        const float yy = -1.0f + 2.0f * (float)h  / 63.0f;
        const float xx = -1.0f + 2.0f * (float)wp / 63.0f;
        const float g  = expf(-2.0f * (yy * yy + xx * xx)) / G;
        const float E  = energy[b * HWW + h * WW + wp];
        sWgt[wave][nloc] = E * g * invS + 1e-10f;
    }

    for (int dy = 0; dy < 7; ++dy) {
        const int hr = min(max(h + dy - RR, 0), HH - 1);
        #pragma unroll
        for (int bt = 0; bt < 2; ++bt) {
            const int wc = min(max(w0 - RR + nloc + bt * 16, 0), WW - 1);
            const size_t baseB = ((size_t)(b * HWW + hr * WW + wc)) * CC;
            v16h b0 = *(const v16h*)(vn + baseB +  0 + hi * 16);
            v16h b1 = *(const v16h*)(vn + baseB + 32 + hi * 16);
            v8f c = {};
            c = WMMA16(a0, b0, c);
            c = WMMA16(a1, b1, c);
            #pragma unroll
            for (int r = 0; r < 8; ++r) sTile[wave][(r + 8 * hi) * 17 + nloc] = c[r];
            #pragma unroll
            for (int d = 0; d < 7; ++d) {            // d = dx + 3
                float v = 0.0f;
                const int n = lane + d - 16 * bt;    // column for pixel m=lane, offset dx=d-3
                if (lane < 16 && n >= 0 && n <= 15)
                    v = sTile[wave][lane * 17 + n] * sWgt[wave][lane];
                v = wave_red(v);
                if (lane == 0) atomicAdd(&sLog[dy * 7 + d], v);
            }
        }
    }
    __syncthreads();
    if (threadIdx.x < KK) atomicAdd(&logits[b * KK + threadIdx.x], sLog[threadIdx.x]);
}

// ---------------------------------------------------------------------------
// Kernel 4: softmax over 49 (with temperature), coarse dx/dy, and the
// 210 -> 128 -> 128 -> 3 GELU MLP head. One block per batch row.
// ---------------------------------------------------------------------------
__global__ __launch_bounds__(128) void k_head(const float* __restrict__ logits,
                                              const float* __restrict__ rpool,
                                              const float* __restrict__ vpool,
                                              const int*   __restrict__ band_idx,
                                              const float* __restrict__ band_table,
                                              const float* __restrict__ logT,
                                              const float* __restrict__ W1, const float* __restrict__ b1,
                                              const float* __restrict__ W2, const float* __restrict__ b2,
                                              const float* __restrict__ W3, const float* __restrict__ b3,
                                              float* __restrict__ out) {
    __shared__ float feat[210];
    __shared__ float h1s[128];
    __shared__ float h2s[128];
    __shared__ float co[2];
    const int b  = blockIdx.x;
    const int tx = threadIdx.x;

    if (tx == 0) {
        const float temp = fmaxf(expf(logT[0]), 1e-3f);
        const float sc   = 0.125f / temp;   // 1/sqrt(C) / temp
        float l[KK];
        float mx = -1e30f;
        for (int i = 0; i < KK; ++i) { l[i] = logits[b * KK + i] * sc; mx = fmaxf(mx, l[i]); }
        float s = 0.0f;
        for (int i = 0; i < KK; ++i) { l[i] = expf(l[i] - mx); s += l[i]; }
        const float r = 1.0f / s;
        float cdx = 0.0f, cdy = 0.0f;
        for (int i = 0; i < KK; ++i) {
            const float p = l[i] * r;
            cdx += p * (float)((i % 7) - RR);
            cdy += p * (float)((i / 7) - RR);
        }
        co[0] = cdx; co[1] = cdy;
        feat[192] = cdx; feat[193] = cdy;
    }
    if (tx < 64) {
        const float rp = rpool[b * CC + tx];
        const float vp = vpool[b * CC + tx];
        feat[tx]        = rp;
        feat[64 + tx]   = vp;
        feat[128 + tx]  = rp * vp;
    }
    if (tx < 16) feat[194 + tx] = band_table[band_idx[b] * 16 + tx];
    __syncthreads();

    float acc = b1[tx];
    for (int i = 0; i < 210; ++i) acc += feat[i] * W1[i * 128 + tx];
    h1s[tx] = gelu_exact(acc);
    __syncthreads();

    acc = b2[tx];
    for (int i = 0; i < 128; ++i) acc += h1s[i] * W2[i * 128 + tx];
    h2s[tx] = gelu_exact(acc);
    __syncthreads();

    if (tx < 3) {
        float r = b3[tx];
        for (int i = 0; i < 128; ++i) r += h2s[i] * W3[i * 3 + tx];
        if (tx == 0) r += co[0];
        else if (tx == 1) r += co[1];
        out[b * 3 + tx] = r;
    }
}

// ---------------------------------------------------------------------------
// Host launcher
// ---------------------------------------------------------------------------
extern "C" void kernel_launch(void* const* d_in, const int* in_sizes, int n_in,
                              void* d_out, int out_size, void* d_ws, size_t ws_size,
                              hipStream_t stream) {
    const float* rfeat  = (const float*)d_in[0];
    const float* vfeat  = (const float*)d_in[1];
    const int*   bandi  = (const int*)  d_in[2];
    const float* Wr     = (const float*)d_in[3];
    const float* Wv     = (const float*)d_in[4];
    const float* logT   = (const float*)d_in[5];
    const float* btab   = (const float*)d_in[6];
    const float* W1     = (const float*)d_in[7];
    const float* b1     = (const float*)d_in[8];
    const float* W2     = (const float*)d_in[9];
    const float* b2     = (const float*)d_in[10];
    const float* W3     = (const float*)d_in[11];
    const float* b3     = (const float*)d_in[12];
    float* out          = (float*)d_out;

    // Workspace layout (bytes; total ~68.2 MB)
    char* ws = (char*)d_ws;
    const size_t SZ_HALF = (size_t)BB * HWW * CC * sizeof(_Float16);   // 33,554,432
    _Float16* rn     = (_Float16*)(ws);
    _Float16* vn     = (_Float16*)(ws + SZ_HALF);
    float*    energy = (float*)   (ws + 2 * SZ_HALF);                          // 1 MB
    float*    sumEG  = (float*)   (ws + 2 * SZ_HALF + 1048576);                // 256 B
    float*    logitsW= (float*)   (ws + 2 * SZ_HALF + 1048576 + 256);          // 12,544 B
    float*    gsum   = (float*)   (ws + 2 * SZ_HALF + 1048576 + 256 + 12544);  // 4 B (padded)
    float*    rpool  = (float*)   (ws + 2 * SZ_HALF + 1048576 + 256 + 12544 + 256);
    float*    vpool  = (float*)   (ws + 2 * SZ_HALF + 1048576 + 256 + 12544 + 256 + 16384);

    k_init<<<1, 64, 0, stream>>>(gsum, sumEG, logitsW);
    k_proj<<<2048, 256, 0, stream>>>(rfeat, vfeat, Wr, Wv, gsum, rn, vn, energy, sumEG);
    k_pool<<<8192, 128, 0, stream>>>(rfeat, vfeat, rpool, vpool);
    k_corr<<<2048, 256, 0, stream>>>(rn, vn, energy, sumEG, gsum, logitsW);
    k_head<<<64, 128, 0, stream>>>(logitsW, rpool, vpool, bandi, btab, logT,
                                   W1, b1, W2, b2, W3, b3, out);
}
